// SparseGCN_58411555225976
// MI455X (gfx1250) — matrix operand order repaired and checked by reading, not compiled, not run on hardware
//
#include <hip/hip_runtime.h>

typedef __attribute__((ext_vector_type(16))) _Float16 v16h;
typedef __attribute__((ext_vector_type(8)))  _Float16 v8h;
typedef __attribute__((ext_vector_type(4)))  _Float16 v4h;
typedef __attribute__((ext_vector_type(8)))  float    v8f;
typedef __attribute__((ext_vector_type(4)))  float    f4;
typedef __attribute__((ext_vector_type(4)))  int      v4i;

#define N_NODES 8192
#define NUM_GRAPHS 64
#define F_IN 128
#define F_HID 256
#define F_OUT 10

// CDNA5 async global->LDS copy path (probe-confirmed live on this toolchain).
#if defined(__has_builtin)
#  if __has_builtin(__builtin_amdgcn_global_load_async_to_lds_b128)
#    define USE_ASYNC_LDS 1
#  endif
#  if __has_builtin(__builtin_amdgcn_s_wait_asynccnt)
#    define WAIT_ASYNC() __builtin_amdgcn_s_wait_asynccnt(0)
#  else
#    define WAIT_ASYNC() asm volatile("s_wait_asynccnt 0" ::: "memory")
#  endif
#endif

// ---------------------------------------------------------------------------
// Prep A: dclamp + W1hT[n][k] = f16(W1[k][n])
// ---------------------------------------------------------------------------
__global__ __launch_bounds__(256) void gcn_prep_small(
    const float* __restrict__ d_inv_sqrt, const float* __restrict__ W1,
    float* __restrict__ dclamp, _Float16* __restrict__ W1hT)
{
  int i = blockIdx.x * 256 + threadIdx.x;
  if (i < N_NODES) {
    float v = d_inv_sqrt[i];
    dclamp[i] = __builtin_isinf(v) ? 0.f : v;
  }
  if (i < F_IN * F_HID) {
    int k = i % F_IN, n = i / F_IN;
    W1hT[i] = (_Float16)W1[(size_t)k * F_HID + n];
  }
}

// ---------------------------------------------------------------------------
// Prep B: XsT[c][r] = f16(d[r] * features[r][c]) via LDS-tiled transpose
// ---------------------------------------------------------------------------
__global__ __launch_bounds__(256) void gcn_prep_xt(
    const float* __restrict__ features, const float* __restrict__ d_inv_sqrt,
    _Float16* __restrict__ XsT)
{
  __shared__ _Float16 t[F_IN][33];          // padded vs bank conflicts
  const int r0 = blockIdx.x * 32;
  const int tid = threadIdx.x;
  for (int idx = tid; idx < 32 * F_IN; idx += 256) {
    int rr = idx / F_IN, c = idx % F_IN;
    float v = d_inv_sqrt[r0 + rr];
    v = __builtin_isinf(v) ? 0.f : v;
    t[c][rr] = (_Float16)(v * features[(size_t)(r0 + rr) * F_IN + c]);
  }
  __syncthreads();
  for (int idx = tid; idx < 32 * F_IN; idx += 256) {
    int c = idx / 32, rr = idx % 32;
    XsT[(size_t)c * N_NODES + r0 + rr] = t[c][rr];
  }
}

// ---------------------------------------------------------------------------
// Double-buffered WMMA GEMM: C[32 x NB] per block = A(fp32->f16) @ BT^T
//   A:  [M][lda] fp32 row-major, streamed non-temporally
//   BT: [NB][KTOT] f16 K-contiguous -> async b128 LDS staging
// Block: 256 threads = 8 waves as 2(M) x 4(N); BK=32; LDS ping-pong.
// Grid: M/32 = 256 blocks (fills all WGPs).
// EPI 0: outF[row][col]  = acc * d[row]
// EPI 1: outHT[col][row] = f16(d[row]*relu(acc + bias[col]))
// EPI 2: x[row][0..9]    = (acc*d[row]) @ W2 + b2   (LDS tile, 4 phases)
// ---------------------------------------------------------------------------
template<int NB, int KTOT, int EPI>
__global__ __launch_bounds__(256) void gcn_gemm(
    const float* __restrict__ A, int lda,
    const _Float16* __restrict__ BT,
    const float* __restrict__ dvec,
    float* __restrict__ outF,
    _Float16* __restrict__ outHT,
    const float* __restrict__ bias,
    const float* __restrict__ W2,
    const float* __restrict__ b2,
    float* __restrict__ xout)
{
  constexpr int NT  = NB / 64;              // WMMA n-tiles per wave (2 or 4)
  constexpr int BCH = NB / 64;              // 16B B-chunks per thread per tile
  constexpr int ASZ = 32 * 32;              // halves per A buffer
  constexpr int XSZ = NB * 32;              // halves per B buffer
  constexpr int SMEM = 2 * (ASZ + XSZ) * 2;
  __shared__ __align__(16) char smem[SMEM];
  _Float16* As = (_Float16*)smem;                         // [2][32][32]
  _Float16* XT = (_Float16*)(smem + 2 * ASZ * 2);         // [2][NB][32]

  const int tid  = threadIdx.x;
  const int lane = tid & 31;
  const int wave = tid >> 5;
  const int wm   = wave & 1;                // 0..1 (M)
  const int wn   = wave >> 1;               // 0..3 (N)
  const int m0   = blockIdx.x * 32;
  const int hi   = lane >> 4;
  const int ln   = lane & 15;

  const int ar = tid >> 3;                  // A stage: row 0..31
  const int ac = (tid & 7) * 4;             // A stage: col 0,4,..,28

  v8f acc[NT];
  #pragma unroll
  for (int t = 0; t < NT; ++t)
    #pragma unroll
    for (int j = 0; j < 8; ++j) acc[t][j] = 0.f;

  // --- B tile stage: contiguous async copy BT[n][k0..k0+32) -> XT[buf][n][.]
  auto stageB = [&](int k0s, int buf) {
    #pragma unroll
    for (int j = 0; j < BCH; ++j) {
      const int ch = tid + j * 256;
      const int n = ch >> 2, koff = (ch & 3) * 8;
      const _Float16* src = BT + (size_t)n * KTOT + k0s + koff;
      _Float16* dst = &XT[buf * XSZ + n * 32 + koff];
#ifdef USE_ASYNC_LDS
      __builtin_amdgcn_global_load_async_to_lds_b128(
          (v4i*)(void*)src, (v4i*)(void*)dst, 0, 0);
#else
      *(v8h*)dst = *(const v8h*)src;
#endif
    }
  };
  auto cvtStoreA = [&](f4 f0, int buf) {
    v4h h4;
    h4[0] = (_Float16)f0.x; h4[1] = (_Float16)f0.y;
    h4[2] = (_Float16)f0.z; h4[3] = (_Float16)f0.w;
    *(v4h*)&As[buf * ASZ + ar * 32 + ac] = h4;
  };

  // --- prologue: tile 0 into buffer 0
  {
    const float* ap = A + (size_t)(m0 + ar) * lda + ac;
    f4 f0 = __builtin_nontemporal_load((const f4*)ap);
    cvtStoreA(f0, 0);
    stageB(0, 0);
#ifdef USE_ASYNC_LDS
    WAIT_ASYNC();
#endif
  }
  __syncthreads();

  int cur = 0;
  for (int k0 = 0; k0 < KTOT; k0 += 32) {
    const int nxt = cur ^ 1;
    const bool more = (k0 + 32) < KTOT;

    // --- issue next tile's global loads (overlap with WMMA below)
    f4 f0{};
#ifndef USE_ASYNC_LDS
    v8h breg[BCH];
#endif
    if (more) {
      const float* ap = A + (size_t)(m0 + ar) * lda + (k0 + 32) + ac;
      f0 = __builtin_nontemporal_load((const f4*)ap);
      if (k0 + 64 < KTOT) __builtin_prefetch(ap + 32, 0, 1);
#ifdef USE_ASYNC_LDS
      stageB(k0 + 32, nxt);
#else
      #pragma unroll
      for (int j = 0; j < BCH; ++j) {
        const int ch = tid + j * 256;
        const int n = ch >> 2, koff = (ch & 3) * 8;
        breg[j] = *(const v8h*)(BT + (size_t)n * KTOT + (k0 + 32) + koff);
      }
#endif
    }

    // --- compute on current buffer (2-deep pipelined B fragments)
    const _Float16* Asc = As + cur * ASZ;
    const _Float16* XTc = XT + cur * XSZ;
    auto loadB = [&](int t) {
      const int col = wn * (NT * 16) + t * 16 + ln;
      v8h lo = *(const v8h*)&XTc[col * 32 + 8 * hi];
      v8h hh = *(const v8h*)&XTc[col * 32 + 16 + 8 * hi];
      v16h b;
      #pragma unroll
      for (int j = 0; j < 8; ++j) { b[j] = lo[j]; b[8 + j] = hh[j]; }
      return b;
    };
    v16h afrag;
    {
      const int row = wm * 16 + ln;
      v8h lo = *(const v8h*)&Asc[row * 32 + 8 * hi];        // K = 8*hi..
      v8h hh = *(const v8h*)&Asc[row * 32 + 16 + 8 * hi];   // K = 16+8*hi..
      #pragma unroll
      for (int j = 0; j < 8; ++j) { afrag[j] = lo[j]; afrag[8 + j] = hh[j]; }
    }
    v16h bf[2];
    bf[0] = loadB(0);
    #pragma unroll
    for (int t = 0; t < NT; ++t) {
      if (t + 1 < NT) bf[(t + 1) & 1] = loadB(t + 1);
      acc[t] = __builtin_amdgcn_wmma_f32_16x16x32_f16(
          false, afrag, false, bf[t & 1], (short)0, acc[t], false, false);
    }

    // --- commit next tile to LDS
    if (more) {
      cvtStoreA(f0, nxt);
#ifndef USE_ASYNC_LDS
      #pragma unroll
      for (int j = 0; j < BCH; ++j) {
        const int ch = tid + j * 256;
        const int n = ch >> 2, koff = (ch & 3) * 8;
        *(v8h*)&XT[nxt * XSZ + n * 32 + koff] = breg[j];
      }
#else
      WAIT_ASYNC();
#endif
    }
    __syncthreads();
    cur = nxt;
  }

  // --- epilogues ------------------------------------------------------------
  if constexpr (EPI == 0) {
    #pragma unroll
    for (int t = 0; t < NT; ++t) {
      const int col = wn * (NT * 16) + t * 16 + ln;
      #pragma unroll
      for (int v = 0; v < 8; ++v) {
        const int grow = m0 + wm * 16 + v + 8 * hi;
        outF[(size_t)grow * NB + col] = acc[t][v] * dvec[grow];
      }
    }
  } else if constexpr (EPI == 1) {
    #pragma unroll
    for (int t = 0; t < NT; ++t) {
      const int col = wn * (NT * 16) + t * 16 + ln;
      const float bb = bias[col];
      #pragma unroll
      for (int v = 0; v < 8; ++v) {
        const int grow = m0 + wm * 16 + v + 8 * hi;
        float val = fmaxf(acc[t][v] + bb, 0.f) * dvec[grow];
        outHT[(size_t)col * N_NODES + grow] = (_Float16)val;   // transposed
      }
    }
  } else {
    // 4 phases: waves wn==ph dump their 32x64 sub-tile (scaled by d) to LDS,
    // then 320 scalar dot products (len 64) accumulate the W2 classifier.
    float* tile = (float*)smem;                 // [32][64] fp32 (reuses stage)
    float s0 = 0.f, s1 = 0.f;
    #pragma unroll
    for (int ph = 0; ph < 4; ++ph) {
      __syncthreads();
      if (wn == ph) {
        #pragma unroll
        for (int t = 0; t < NT; ++t)
          #pragma unroll
          for (int v = 0; v < 8; ++v) {
            const int rl = wm * 16 + v + 8 * hi;
            tile[rl * 64 + t * 16 + ln] = acc[t][v] * dvec[m0 + rl];
          }
      }
      __syncthreads();
      int oi = tid;                             // outputs 0..319 (32 rows x 10)
      {
        int row = oi / 10, c = oi % 10;
        float ss = 0.f;
        for (int k = 0; k < 64; ++k) ss += tile[row * 64 + k] * W2[(ph * 64 + k) * 10 + c];
        s0 += ss;
      }
      oi = tid + 256;
      if (oi < 320) {
        int row = oi / 10, c = oi % 10;
        float ss = 0.f;
        for (int k = 0; k < 64; ++k) ss += tile[row * 64 + k] * W2[(ph * 64 + k) * 10 + c];
        s1 += ss;
      }
    }
    int oi = tid;
    if (oi < 320) xout[(size_t)(m0 + oi / 10) * 10 + oi % 10] = s0 + b2[oi % 10];
    oi = tid + 256;
    if (oi < 320) xout[(size_t)(m0 + oi / 10) * 10 + oi % 10] = s1 + b2[oi % 10];
  }
}

// ---------------------------------------------------------------------------
// Deterministic per-graph mean (tree reduction, no float atomics)
// ---------------------------------------------------------------------------
__global__ __launch_bounds__(256) void gcn_segmean(
    const float* __restrict__ x, const int* __restrict__ gid,
    float* __restrict__ out)
{
  const int g = blockIdx.x;
  const int t = threadIdx.x;
  float ps[F_OUT];
  #pragma unroll
  for (int c = 0; c < F_OUT; ++c) ps[c] = 0.f;
  float cnt = 0.f;
  for (int r = t; r < N_NODES; r += 256) {
    if (gid[r] == g) {
      cnt += 1.f;
      #pragma unroll
      for (int c = 0; c < F_OUT; ++c) ps[c] += x[(size_t)r * F_OUT + c];
    }
  }
  __shared__ float red[256];
  __shared__ float sums[F_OUT];
  __shared__ float total;
  red[t] = cnt;
  __syncthreads();
  for (int s = 128; s > 0; s >>= 1) { if (t < s) red[t] += red[t + s]; __syncthreads(); }
  if (t == 0) total = red[0];
  __syncthreads();
  for (int c = 0; c < F_OUT; ++c) {
    red[t] = ps[c];
    __syncthreads();
    for (int s = 128; s > 0; s >>= 1) { if (t < s) red[t] += red[t + s]; __syncthreads(); }
    if (t == 0) sums[c] = red[0];
    __syncthreads();
  }
  if (t < F_OUT) out[g * F_OUT + t] = sums[t] / fmaxf(total, 1.f);
}

// ---------------------------------------------------------------------------
extern "C" void kernel_launch(void* const* d_in, const int* in_sizes, int n_in,
                              void* d_out, int out_size, void* d_ws, size_t ws_size,
                              hipStream_t stream) {
  const float* adj  = (const float*)d_in[0];   // [8192][8192]
  const float* feat = (const float*)d_in[1];   // [8192][128]
  const float* dis  = (const float*)d_in[2];   // [8192]
  const float* W1   = (const float*)d_in[3];   // [128][256]
  const float* b1   = (const float*)d_in[4];   // [256]
  const float* W2   = (const float*)d_in[5];   // [256][10]
  const float* b2   = (const float*)d_in[6];   // [10]
  const int*   gid  = (const int*)d_in[7];     // [8192]
  float* out = (float*)d_out;                  // [64][10]
  (void)in_sizes; (void)n_in; (void)out_size; (void)ws_size;

  char* ws = (char*)d_ws;
  size_t off = 0;
  float*    dclamp = (float*)(ws + off);    off += (size_t)N_NODES * 4;
  _Float16* XsT    = (_Float16*)(ws + off); off += (size_t)F_IN * N_NODES * 2;   // [128][8192]
  _Float16* W1hT   = (_Float16*)(ws + off); off += (size_t)F_HID * F_IN * 2;     // [256][128]
  float*    T1     = (float*)(ws + off);    off += (size_t)N_NODES * F_IN * 4;   // [8192][128]
  _Float16* hsT    = (_Float16*)(ws + off); off += (size_t)F_HID * N_NODES * 2;  // [256][8192]
  float*    x      = (float*)(ws + off);                                         // [8192][10]

  gcn_prep_small<<<(F_IN * F_HID) / 256, 256, 0, stream>>>(dis, W1, dclamp, W1hT);
  gcn_prep_xt<<<N_NODES / 32, 256, 0, stream>>>(feat, dis, XsT);

  // T1 = d ∘ (A @ Xs)            M=8192 N=128 K=8192
  gcn_gemm<128, 8192, 0><<<N_NODES / 32, 256, 0, stream>>>(
      adj, N_NODES, XsT, dclamp, T1, nullptr, nullptr, nullptr, nullptr, nullptr);

  // hsT = f16(d ∘ relu(T1 @ W1 + b1))^T   M=8192 N=256 K=128
  gcn_gemm<256, 128, 1><<<N_NODES / 32, 256, 0, stream>>>(
      T1, F_IN, W1hT, dclamp, nullptr, hsT, b1, nullptr, nullptr, nullptr);

  // x = (d ∘ (A @ hs)) @ W2 + b2          M=8192 N=256 K=8192
  gcn_gemm<256, 8192, 2><<<N_NODES / 32, 256, 0, stream>>>(
      adj, N_NODES, hsT, dclamp, nullptr, nullptr, nullptr, W2, b2, x);

  gcn_segmean<<<NUM_GRAPHS, 256, 0, stream>>>(x, gid, out);
}
